// ScaledAttention_43757126812299
// MI455X (gfx1250) — compile-verified, matrix operand order checked by reading, not run on hardware
//
#include <hip/hip_runtime.h>

typedef __attribute__((ext_vector_type(16))) _Float16 v16h;
typedef __attribute__((ext_vector_type(8)))  float    v8f;
typedef __attribute__((ext_vector_type(4)))  float    v4f;

#define BATCH 16
#define SEQ   2048
#define DIM   128
#define TILE_M 16
#define NWAVES 8   // 256 threads = 8 wave32

// Async-DMA a contiguous 16 KB block (one V k-tile: 32 rows x 128 f32) from
// global memory into LDS: 256 threads x 4 x b128. The instruction offset is
// added to BOTH the LDS and global addresses (CDNA5 ISA),
// which is exactly right for a linear copy. Tracked by ASYNCcnt.
__device__ __forceinline__ void async_copy_16k(unsigned lds_off, const float* gsrc) {
    const unsigned t = (unsigned)threadIdx.x * 16u;
    const unsigned l = lds_off + t;
    const unsigned long long g = (unsigned long long)(uintptr_t)gsrc + t;
    asm volatile(
        "global_load_async_to_lds_b128 %0, %1, off\n\t"
        "global_load_async_to_lds_b128 %0, %1, off offset:4096\n\t"
        "global_load_async_to_lds_b128 %0, %1, off offset:8192\n\t"
        "global_load_async_to_lds_b128 %0, %1, off offset:12288"
        :: "v"(l), "v"(g)
        : "memory");
}

// Build a 16-half A/B fragment from 32 consecutive-K f32 values laid out per
// CDNA5 ISA §7.12.2: halves 0-7 <- p[0..7], halves 8-15 <- p[16..23]
// (the caller folds the +8*hi lane-half K offset into p). Vectorized b128 loads.
__device__ __forceinline__ v16h cvt_frag(const float* p) {
    const v4f a = *(const v4f*)(p);
    const v4f b = *(const v4f*)(p + 4);
    const v4f c = *(const v4f*)(p + 16);
    const v4f d = *(const v4f*)(p + 20);
    v16h f;
    f[0]  = (_Float16)a.x; f[1]  = (_Float16)a.y; f[2]  = (_Float16)a.z; f[3]  = (_Float16)a.w;
    f[4]  = (_Float16)b.x; f[5]  = (_Float16)b.y; f[6]  = (_Float16)b.z; f[7]  = (_Float16)b.w;
    f[8]  = (_Float16)c.x; f[9]  = (_Float16)c.y; f[10] = (_Float16)c.z; f[11] = (_Float16)c.w;
    f[12] = (_Float16)d.x; f[13] = (_Float16)d.y; f[14] = (_Float16)d.z; f[15] = (_Float16)d.w;
    return f;
}

__global__ __launch_bounds__(256)
void attn_fused_kernel(const float* __restrict__ qg,
                       const float* __restrict__ kg,
                       const float* __restrict__ vg,
                       float* __restrict__ outg,
                       float* __restrict__ attng) {
    __shared__ float S[TILE_M][SEQ];   // 128 KB: scores -> probabilities
    __shared__ float Vt[2][32][DIM];   // 32 KB: double-buffered V k-tiles

    const int wg   = blockIdx.x;        // 0..2047
    const int b    = wg >> 7;
    const int m0   = (wg & 127) * TILE_M;

    const int tid  = threadIdx.x;
    const int wave = tid >> 5;
    const int lane = tid & 31;
    const int hi   = lane >> 4;         // 0: lanes 0-15, 1: lanes 16-31
    const int ln   = lane & 15;

    const float* Qb = qg + ((size_t)b * SEQ + m0) * DIM;
    const float* Kb = kg + (size_t)b * SEQ * DIM;
    const float* Vb = vg + (size_t)b * SEQ * DIM;

    // Low 32 bits of the flat LDS pointer == LDS byte offset.
    const unsigned vbase = (unsigned)(uintptr_t)&Vt[0][0][0];

    // Kick off the DMA of V tile 0 now; phases 1-2 fully hide its latency.
    async_copy_16k(vbase, Vb);

    // ---------------- Phase 1: S = Q @ K^T (f16 WMMA, f32 accum) -------------
    v16h qa[4];
#pragma unroll
    for (int kt = 0; kt < 4; ++kt)
        qa[kt] = cvt_frag(Qb + ln * DIM + kt * 32 + hi * 8);

    for (int nt = wave; nt < SEQ / 16; nt += NWAVES) {
        const int n0 = nt * 16;
        const float* kp = Kb + (size_t)(n0 + ln) * DIM + hi * 8;
        __builtin_prefetch(kp + (size_t)NWAVES * 16 * DIM, 0, 3);  // next K tile
        v8f c = {};
#pragma unroll
        for (int kt = 0; kt < 4; ++kt) {
            const v16h bfrag = cvt_frag(kp + kt * 32);
            c = __builtin_amdgcn_wmma_f32_16x16x32_f16(
                    false, qa[kt], false, bfrag, (short)0, c, false, false);
        }
#pragma unroll
        for (int r = 0; r < 8; ++r)
            S[r + 8 * hi][n0 + ln] = c[r];   // C: VGPR r -> (M = r+8*hi, N = ln)
    }
    __syncthreads();

    // ---------------- Phase 2: row softmax + stream attn to HBM --------------
#pragma unroll 1
    for (int rr = 0; rr < 2; ++rr) {
        const int row = wave * 2 + rr;
        float mx = -3.402823466e38f;
        v4f sv[16];
#pragma unroll
        for (int j = 0; j < 16; ++j) {
            sv[j] = *(const v4f*)&S[row][4 * lane + 128 * j];
            mx = fmaxf(mx, fmaxf(fmaxf(sv[j].x, sv[j].y), fmaxf(sv[j].z, sv[j].w)));
        }
#pragma unroll
        for (int off = 16; off > 0; off >>= 1)
            mx = fmaxf(mx, __shfl_xor(mx, off, 32));

        float sum = 0.f;
#pragma unroll
        for (int j = 0; j < 16; ++j) {
            sv[j].x = __expf(sv[j].x - mx);
            sv[j].y = __expf(sv[j].y - mx);
            sv[j].z = __expf(sv[j].z - mx);
            sv[j].w = __expf(sv[j].w - mx);
            sum += (sv[j].x + sv[j].y) + (sv[j].z + sv[j].w);
        }
#pragma unroll
        for (int off = 16; off > 0; off >>= 1)
            sum += __shfl_xor(sum, off, 32);
        const float inv = 1.f / sum;

        float* arow = attng + (size_t)(b * SEQ + m0 + row) * SEQ;
#pragma unroll
        for (int j = 0; j < 16; ++j) {
            v4f p;
            p.x = sv[j].x * inv; p.y = sv[j].y * inv;
            p.z = sv[j].z * inv; p.w = sv[j].w * inv;
            *(v4f*)&S[row][4 * lane + 128 * j] = p;              // P for phase 3
            // attn is a 268 MB write-once stream: non-temporal, keep L2 for K/V
            __builtin_nontemporal_store(p, (v4f*)&arow[4 * lane + 128 * j]);
        }
    }
    __syncthreads();

    // -------- Phase 3: O = P @ V with async double-buffered V staging --------
    {
        const int n0 = wave * 16;
        v8f c = {};
#pragma unroll 1
        for (int kt = 0; kt < 64; ++kt) {
            const int cur = kt & 1;
            // Prefetch next tile into the idle buffer (wraps to a harmless
            // dummy copy on the last iteration so the wait count is uniform).
            const int knext = (kt + 1) & 63;
            async_copy_16k(vbase + (unsigned)(cur ^ 1) * 16384u,
                           Vb + (size_t)knext * 32 * DIM);
            // Oldest 4 ASYNC ops (= tile kt) complete in order; allow the 4
            // just-issued for tile kt+1 to stay in flight.
            asm volatile("s_wait_asynccnt 0x4" ::: "memory");
            __syncthreads();               // all waves' DMAs for tile kt visible

            const v16h afrag = cvt_frag(&S[ln][kt * 32 + hi * 8]);
            const float* vt = &Vt[cur][hi * 8][n0 + ln];   // B[k,n] = V[k0+k, n]
            v16h bfrag;
#pragma unroll
            for (int h = 0; h < 8; ++h) bfrag[h]     = (_Float16)vt[h * DIM];
#pragma unroll
            for (int h = 0; h < 8; ++h) bfrag[h + 8] = (_Float16)vt[(h + 16) * DIM];
            c = __builtin_amdgcn_wmma_f32_16x16x32_f16(
                    false, afrag, false, bfrag, (short)0, c, false, false);
            __syncthreads();               // reads done before buffer is reused
        }
        float* op = outg + ((size_t)b * SEQ + m0) * DIM;
#pragma unroll
        for (int r = 0; r < 8; ++r)
            op[(size_t)(r + 8 * hi) * DIM + n0 + ln] = c[r];
    }
}

extern "C" void kernel_launch(void* const* d_in, const int* in_sizes, int n_in,
                              void* d_out, int out_size, void* d_ws, size_t ws_size,
                              hipStream_t stream) {
    const float* q = (const float*)d_in[0];
    const float* k = (const float*)d_in[1];
    const float* v = (const float*)d_in[2];
    float* out  = (float*)d_out;                       // [16,2048,128]
    float* attn = out + (size_t)BATCH * SEQ * DIM;     // [16,2048,2048]

    dim3 grid(BATCH * (SEQ / TILE_M));   // 2048 workgroups
    dim3 block(NWAVES * 32);             // 8 wave32
    hipLaunchKernelGGL(attn_fused_kernel, grid, block, 0, stream,
                       q, k, v, out, attn);
}
